// Network_13374528159896
// MI455X (gfx1250) — matrix operand order repaired and checked by reading, not compiled, not run on hardware
//
#include <hip/hip_runtime.h>

#define NN   50000
#define EE   800000
#define IND  64
#define HH   128
#define LLAY 3
#define PPRI 4
#define OUTD 2

typedef __attribute__((ext_vector_type(16))) _Float16 v16h;
typedef __attribute__((ext_vector_type(8)))  float    v8f;

enum { EPI_STORE = 0, EPI_ELU = 1, EPI_RELU_JK = 2, EPI_LA = 3, EPI_RELU = 4 };
enum { ASRC_PLAIN = 0, ASRC_MEAN = 1, ASRC_ADD = 2 };

// ---------------------------------------------------------------------------
// Tiny kernels: softmax of architecture alphas, f32 -> f16 transposed weights
// ---------------------------------------------------------------------------
__global__ void prep_weights_kernel(const float* __restrict__ naa,
                                    const float* __restrict__ sca,
                                    const float* __restrict__ laa,
                                    float* __restrict__ naw,
                                    float* __restrict__ scw,
                                    float* __restrict__ law) {
  int t = threadIdx.x;
  if (t < 9) {                       // na_alphas [3,3,4] softmax over P
    const float* a = naa + t * 4;
    float m = fmaxf(fmaxf(a[0], a[1]), fmaxf(a[2], a[3]));
    float e[4]; float s = 0.f;
    for (int j = 0; j < 4; ++j) { e[j] = expf(a[j] - m); s += e[j]; }
    for (int j = 0; j < 4; ++j) naw[t * 4 + j] = e[j] / s;
  } else if (t < 11) {               // sc_alphas [2,2]
    int r = t - 9;
    const float* a = sca + r * 2;
    float m = fmaxf(a[0], a[1]);
    float e0 = expf(a[0] - m), e1 = expf(a[1] - m);
    scw[r * 2 + 0] = e0 / (e0 + e1);
    scw[r * 2 + 1] = e1 / (e0 + e1);
  } else if (t == 11) {              // la_alphas [1,3]
    float m = fmaxf(fmaxf(laa[0], laa[1]), laa[2]);
    float e0 = expf(laa[0] - m), e1 = expf(laa[1] - m), e2 = expf(laa[2] - m);
    float s = e0 + e1 + e2;
    law[0] = e0 / s; law[1] = e1 / s; law[2] = e2 / s;
  }
}

// dst is [nmat][128][K] (transposed), src is [nmat][K][128]
__global__ void cvtT_kernel(const float* __restrict__ src, _Float16* __restrict__ dst,
                            int K_, long long total) {
  long long i = (long long)blockIdx.x * blockDim.x + threadIdx.x;
  if (i >= total) return;
  long long per = (long long)K_ * 128;
  long long m = i / per, r = i % per;
  int n = (int)(r / K_), k = (int)(r % K_);
  dst[m * per + (long long)n * K_ + k] = (_Float16)src[m * per + (long long)k * 128 + n];
}

// ---------------------------------------------------------------------------
// Graph kernels
// ---------------------------------------------------------------------------
__global__ void deg_count_kernel(const int* __restrict__ src, const int* __restrict__ dst,
                                 float* __restrict__ dout, float* __restrict__ din, int E_) {
  int e = blockIdx.x * blockDim.x + threadIdx.x;
  if (e >= E_) return;
  atomicAdd(&dout[src[e]], 1.0f);
  atomicAdd(&din[dst[e]], 1.0f);
}

__global__ void deg_recip_kernel(float* __restrict__ d, int n) {
  int i = blockIdx.x * blockDim.x + threadIdx.x;
  if (i >= n) return;
  d[i] = 1.0f / fmaxf(d[i], 1.0f);
}

__global__ void norm_kernel(const int* __restrict__ src, const int* __restrict__ dst,
                            const float* __restrict__ rdo, const float* __restrict__ rdi,
                            float* __restrict__ nrm, int E_) {
  int e = blockIdx.x * blockDim.x + threadIdx.x;
  if (e >= E_) return;
  nrm[e] = sqrtf(rdo[src[e]] * rdi[dst[e]]);   // 1/sqrt(deg_out*deg_in)
}

// one wave = one edge (32 lanes x 4 floats = 128 feats); sum_agg + gcn_agg fused
__global__ void scatter_kernel(const float* __restrict__ h,
                               const int* __restrict__ src, const int* __restrict__ dst,
                               const float* __restrict__ nrm,
                               float* __restrict__ sum_agg, float* __restrict__ gcn_agg,
                               int E_) {
  long long idx = (long long)blockIdx.x * blockDim.x + threadIdx.x;
  if (idx >= (long long)E_ * 32) return;
  int e = (int)(idx >> 5);
  int c = (int)(idx & 31) << 2;
  int s = src[e], d = dst[e];
  float nv = nrm[e];
  float4 xv = *(const float4*)(h + (size_t)s * HH + c);
  float* sp = sum_agg + (size_t)d * HH + c;
  float* gp = gcn_agg + (size_t)d * HH + c;
  atomicAdd(sp + 0, xv.x); atomicAdd(sp + 1, xv.y);
  atomicAdd(sp + 2, xv.z); atomicAdd(sp + 3, xv.w);
  atomicAdd(gp + 0, xv.x * nv); atomicAdd(gp + 1, xv.y * nv);
  atomicAdd(gp + 2, xv.z * nv); atomicAdd(gp + 3, xv.w * nv);
}

// ---------------------------------------------------------------------------
// WMMA GEMM: C[M,128] = act(A[M,K] @ B[K,128] + bias) with fused epilogues.
// A synthesized on the fly (plain / A0 + A1*rdeg / A0 + A1), fp32 -> f16 in LDS.
// B pre-transposed f16 [128][K], staged with GLOBAL_LOAD_ASYNC_TO_LDS_B128.
// Block tile 128x128, 8 waves, wave = 16-row strip across all 8 N-tiles.
// ---------------------------------------------------------------------------
#define LDK 40  // padded LDS K-stride (f16): 80B rows -> 16B-aligned b128, conflict-free

template <int MODE, int ASRC>
__global__ __launch_bounds__(256) void gemm_wmma_kernel(
    const float* __restrict__ A0, const float* __restrict__ A1,
    const float* __restrict__ rdeg,
    const _Float16* __restrict__ B, const float* __restrict__ bias,
    float* __restrict__ out, int ldo, int ocol,
    float* __restrict__ aux0, int aux_ld, int aux_col,   // jk_cat
    float* __restrict__ aux1,                            // jk_sum / s
    const float* __restrict__ w0,                        // scalar coeff(s)
    int M, int K, int accum) {
  __shared__ _Float16 As[128 * LDK];
  __shared__ _Float16 Bs[128 * LDK];

  const int tid = threadIdx.x;
  const int wave = tid >> 5;
  const int lane = tid & 31;
  const int hl = lane >> 4;     // lane half (0/1)
  const int lh = lane & 15;     // lane-in-half
  const int row0 = blockIdx.x * 128;

  v8f c[8];
#pragma unroll
  for (int t = 0; t < 8; ++t)
#pragma unroll
    for (int j = 0; j < 8; ++j) c[t][j] = 0.0f;

  // per-thread staging coordinates (fixed across k-steps)
  const int sm = tid >> 1;             // row (A) / col (B) within tile
  const int skq = (tid & 1) << 4;      // 16-element k chunk
  const unsigned bs_lds = (unsigned)(unsigned long long)&Bs[sm * LDK + skq];

  for (int kt = 0; kt < K; kt += 32) {
    // ---- async-stage B tile: 2 x b128 per lane, global+LDS share IOFFSET ----
    {
      unsigned long long ga = (unsigned long long)(B + (size_t)sm * K + kt + skq);
      asm volatile(
          "global_load_async_to_lds_b128 %0, %1, off\n\t"
          "global_load_async_to_lds_b128 %0, %1, off offset:16"
          :: "v"(bs_lds), "v"(ga) : "memory");
    }
    // ---- stage A tile (128x32 fp32 -> f16), thread covers (m, 16 k's) ----
    {
      int gm = row0 + sm;
      float v[16];
      if (gm < M) {
        const float4* pa = (const float4*)(A0 + (size_t)gm * K + kt + skq);
        float4 q[4];
#pragma unroll
        for (int j = 0; j < 4; ++j) q[j] = pa[j];
        if (ASRC != ASRC_PLAIN) {
          float rd = (ASRC == ASRC_MEAN) ? rdeg[gm] : 1.0f;
          const float4* pb = (const float4*)(A1 + (size_t)gm * K + kt + skq);
#pragma unroll
          for (int j = 0; j < 4; ++j) {
            float4 b4 = pb[j];
            q[j].x += b4.x * rd; q[j].y += b4.y * rd;
            q[j].z += b4.z * rd; q[j].w += b4.w * rd;
          }
        }
#pragma unroll
        for (int j = 0; j < 4; ++j) {
          v[4 * j + 0] = q[j].x; v[4 * j + 1] = q[j].y;
          v[4 * j + 2] = q[j].z; v[4 * j + 3] = q[j].w;
        }
      } else {
#pragma unroll
        for (int j = 0; j < 16; ++j) v[j] = 0.0f;
      }
      unsigned int* dp = (unsigned int*)&As[sm * LDK + skq];
#pragma unroll
      for (int j = 0; j < 8; ++j) {
        union { _Float16 h[2]; unsigned int u; } pk;
        pk.h[0] = (_Float16)v[2 * j];
        pk.h[1] = (_Float16)v[2 * j + 1];
        dp[j] = pk.u;
      }
    }
    asm volatile("s_wait_asynccnt 0x0" ::: "memory");
    __syncthreads();

    // ---- fragments + WMMA ----
    // A layout: lane m = lh, k = 16*hl + e (contiguous, matches B)
    int am = wave * 16 + lh;
    v16h a;
    ((uint4*)&a)[0] = *(const uint4*)&As[am * LDK + hl * 16];
    ((uint4*)&a)[1] = *(const uint4*)&As[am * LDK + hl * 16 + 8];
#pragma unroll
    for (int t = 0; t < 8; ++t) {
      // B layout: lane n = lh, k = 16*hl + e (contiguous)
      int bn = t * 16 + lh;
      v16h b;
      ((uint4*)&b)[0] = *(const uint4*)&Bs[bn * LDK + hl * 16];
      ((uint4*)&b)[1] = *(const uint4*)&Bs[bn * LDK + hl * 16 + 8];
      c[t] = __builtin_amdgcn_wmma_f32_16x16x32_f16(
          false, a, false, b, (short)0, c[t], false, false);
    }
    __syncthreads();
  }

  // ---- epilogue: D lane = (n = t*16+lh, m = row0 + wave*16 + 8*hl + v) ----
  float wv0 = w0 ? w0[0] : 1.0f;
#pragma unroll
  for (int t = 0; t < 8; ++t) {
    int n = t * 16 + lh;
    float bv = bias[n];
#pragma unroll
    for (int v = 0; v < 8; ++v) {
      int gm = row0 + wave * 16 + hl * 8 + v;
      if (gm >= M) continue;
      float acc = c[t][v] + bv;
      size_t oidx = (size_t)gm * ldo + ocol + n;
      if (MODE == EPI_STORE) {
        out[oidx] = acc;
      } else if (MODE == EPI_ELU) {
        float ev = acc > 0.0f ? acc : (expf(acc) - 1.0f);
        float r = wv0 * ev;
        out[oidx] = accum ? (out[oidx] + r) : r;
      } else if (MODE == EPI_RELU_JK) {
        float hv = fmaxf(acc, 0.0f);
        out[oidx] = hv;                                   // h for next layer
        float sv = wv0 * hv;                              // sc-scaled jk entry
        aux0[(size_t)gm * aux_ld + aux_col + n] = sv;     // jk_cat
        size_t ji = (size_t)gm * HH + n;
        aux1[ji] = accum ? (aux1[ji] + sv) : sv;          // jk_sum
      } else if (MODE == EPI_LA) {
        float s = aux1[(size_t)gm * HH + n];
        float mg = w0[0] * fmaxf(s, 0.0f)
                 + w0[1] * fmaxf(s * (1.0f / 3.0f), 0.0f)
                 + w0[2] * fmaxf(acc, 0.0f);
        out[oidx] = fmaxf(mg, 0.0f);
      } else {  // EPI_RELU
        out[oidx] = fmaxf(acc, 0.0f);
      }
    }
  }
}

// host-side dispatch over the 7 used (MODE, ASRC) combos
static void launch_gemm(int mode, int asrc, int G, hipStream_t stream,
                        const float* A0, const float* A1, const float* rdeg,
                        const _Float16* B, const float* bias,
                        float* out, int ldo, int ocol,
                        float* aux0, int aux_ld, int aux_col, float* aux1,
                        const float* w0, int M, int K, int accum) {
#define GL(MD, AS)                                                             \
  gemm_wmma_kernel<MD, AS><<<G, 256, 0, stream>>>(                             \
      A0, A1, rdeg, B, bias, out, ldo, ocol, aux0, aux_ld, aux_col, aux1, w0,  \
      M, K, accum)
  if (mode == EPI_STORE)        GL(EPI_STORE, ASRC_PLAIN);
  else if (mode == EPI_ELU) {
    if (asrc == ASRC_PLAIN)     GL(EPI_ELU, ASRC_PLAIN);
    else if (asrc == ASRC_MEAN) GL(EPI_ELU, ASRC_MEAN);
    else                        GL(EPI_ELU, ASRC_ADD);
  }
  else if (mode == EPI_RELU_JK) GL(EPI_RELU_JK, ASRC_PLAIN);
  else if (mode == EPI_LA)      GL(EPI_LA, ASRC_PLAIN);
  else                          GL(EPI_RELU, ASRC_PLAIN);
#undef GL
}

// final [N,128] @ [128,2] + b
__global__ void clf2_kernel(const float* __restrict__ m1, const float* __restrict__ W2,
                            const float* __restrict__ b2, float* __restrict__ out, int M) {
  int m = blockIdx.x * blockDim.x + threadIdx.x;
  if (m >= M) return;
  float a0 = b2[0], a1 = b2[1];
  const float* row = m1 + (size_t)m * HH;
#pragma unroll 4
  for (int k = 0; k < HH; ++k) {
    float x = row[k];
    a0 += x * W2[2 * k];
    a1 += x * W2[2 * k + 1];
  }
  out[2 * m + 0] = a0;
  out[2 * m + 1] = a1;
}

// ---------------------------------------------------------------------------
extern "C" void kernel_launch(void* const* d_in, const int* in_sizes, int n_in,
                              void* d_out, int out_size, void* d_ws, size_t ws_size,
                              hipStream_t stream) {
  (void)in_sizes; (void)n_in; (void)out_size; (void)ws_size;

  const float* x        = (const float*)d_in[0];
  const int*   ei[3]    = {(const int*)d_in[1], (const int*)d_in[2], (const int*)d_in[3]};
  const float* lin1_W   = (const float*)d_in[4];
  const float* lin1_b   = (const float*)d_in[5];
  const float* lin2_W   = (const float*)d_in[6];
  const float* lin2_b   = (const float*)d_in[7];
  const float* na_W     = (const float*)d_in[8];
  const float* na_b     = (const float*)d_in[9];
  const float* la_cat_W = (const float*)d_in[10];
  const float* la_cat_b = (const float*)d_in[11];
  const float* clf_W1   = (const float*)d_in[12];
  const float* clf_b1   = (const float*)d_in[13];
  const float* clf_W2   = (const float*)d_in[14];
  const float* clf_b2   = (const float*)d_in[15];
  const float* na_alpha = (const float*)d_in[16];
  const float* sc_alpha = (const float*)d_in[17];
  const float* la_alpha = (const float*)d_in[18];

  // ---- workspace layout (~270 MB) ----
  char* base = (char*)d_ws;
  size_t off = 0;
  auto alloc = [&](size_t nbytes) -> void* {
    off = (off + 255) & ~(size_t)255;
    void* p = base + off;
    off += nbytes;
    return p;
  };
  float* h       = (float*)alloc((size_t)NN * HH * 4);
  float* sum_agg = (float*)alloc((size_t)NN * HH * 4);
  float* gcn_agg = (float*)alloc((size_t)NN * HH * 4);
  float* xs_cat  = (float*)alloc((size_t)NN * 3 * HH * 4);
  float* jk_sum  = (float*)alloc((size_t)NN * HH * 4);
  float* jk_cat  = (float*)alloc((size_t)NN * 3 * HH * 4);
  float* merge   = sum_agg;   // aggs dead after last na GEMMs
  float* m1      = gcn_agg;
  float* rdi[3], *rdo[3], *nrm[3];
  for (int r = 0; r < 3; ++r) {
    rdi[r] = (float*)alloc((size_t)NN * 4);
    rdo[r] = (float*)alloc((size_t)NN * 4);
    nrm[r] = (float*)alloc((size_t)EE * 4);
  }
  _Float16* lw1   = (_Float16*)alloc((size_t)IND * HH * 2);
  _Float16* lw2   = (_Float16*)alloc((size_t)3 * HH * HH * 2);
  _Float16* nawT  = (_Float16*)alloc((size_t)36 * HH * HH * 2);
  _Float16* lawT  = (_Float16*)alloc((size_t)3 * HH * HH * 2);
  _Float16* c1wT  = (_Float16*)alloc((size_t)HH * HH * 2);
  float* na_w_d = (float*)alloc(36 * 4);
  float* sc_w_d = (float*)alloc(4 * 4);
  float* la_w_d = (float*)alloc(3 * 4);

  const int G = (NN + 127) / 128;   // GEMM grid (391 blocks)

  // ---- architecture weights + f16 transposed weight matrices ----
  prep_weights_kernel<<<1, 32, 0, stream>>>(na_alpha, sc_alpha, la_alpha,
                                            na_w_d, sc_w_d, la_w_d);
  auto cvt = [&](const float* src, _Float16* dst, int K, long long total) {
    cvtT_kernel<<<(int)((total + 255) / 256), 256, 0, stream>>>(src, dst, K, total);
  };
  cvt(lin1_W,   lw1,  IND,      (long long)IND * HH);
  cvt(lin2_W,   lw2,  3 * HH,   (long long)3 * HH * HH);
  cvt(na_W,     nawT, HH,       (long long)36 * HH * HH);
  cvt(la_cat_W, lawT, 3 * HH,   (long long)3 * HH * HH);
  cvt(clf_W1,   c1wT, HH,       (long long)HH * HH);

  // ---- per-relation degrees + gcn edge norms (edges constant across layers) ----
  for (int r = 0; r < 3; ++r) {
    const int* src = ei[r];
    const int* dst = ei[r] + EE;
    hipMemsetAsync(rdo[r], 0, (size_t)NN * 4, stream);
    hipMemsetAsync(rdi[r], 0, (size_t)NN * 4, stream);
    deg_count_kernel<<<(EE + 255) / 256, 256, 0, stream>>>(src, dst, rdo[r], rdi[r], EE);
    deg_recip_kernel<<<(NN + 255) / 256, 256, 0, stream>>>(rdo[r], NN);
    deg_recip_kernel<<<(NN + 255) / 256, 256, 0, stream>>>(rdi[r], NN);
    norm_kernel<<<(EE + 255) / 256, 256, 0, stream>>>(src, dst, rdo[r], rdi[r], nrm[r], EE);
  }

  // ---- h = x @ lin1_W + b ----
  launch_gemm(EPI_STORE, ASRC_PLAIN, G, stream,
              x, nullptr, nullptr, lw1, lin1_b,
              h, HH, 0, nullptr, 0, 0, nullptr, nullptr, NN, IND, 0);

  // ---- 3 layers ----
  for (int i = 0; i < LLAY; ++i) {
    for (int r = 0; r < 3; ++r) {
      const int* src = ei[r];
      const int* dst = ei[r] + EE;
      hipMemsetAsync(sum_agg, 0, (size_t)NN * HH * 4, stream);
      hipMemsetAsync(gcn_agg, 0, (size_t)NN * HH * 4, stream);
      long long sthreads = (long long)EE * 32;
      scatter_kernel<<<(int)((sthreads + 255) / 256), 256, 0, stream>>>(
          h, src, dst, nrm[r], sum_agg, gcn_agg, EE);

      for (int p = 0; p < PPRI; ++p) {
        const float* A0; const float* A1 = nullptr; int asrc;
        switch (p) {
          case 0: A0 = gcn_agg; asrc = ASRC_PLAIN; break;              // gcn
          case 1: A0 = h; A1 = sum_agg; asrc = ASRC_MEAN; break;       // x + mean
          case 2: A0 = h; A1 = sum_agg; asrc = ASRC_ADD; break;        // x + sum
          default: A0 = sum_agg; asrc = ASRC_PLAIN; break;             // sum
        }
        int mi = (i * 3 + r) * 4 + p;
        launch_gemm(EPI_ELU, asrc, G, stream,
                    A0, A1, rdi[r],
                    nawT + (size_t)mi * HH * HH, na_b + (size_t)mi * HH,
                    xs_cat, 3 * HH, r * HH,
                    nullptr, 0, 0, nullptr, na_w_d + mi,
                    NN, HH, p > 0 ? 1 : 0);
      }
    }
    // h = relu(xs_cat @ lin2_W + b); jk_sum/jk_cat fused (sc-scaled, last sc=1)
    const float* scp = (i < LLAY - 1) ? (sc_w_d + i * 2 + 1) : nullptr;
    launch_gemm(EPI_RELU_JK, ASRC_PLAIN, G, stream,
                xs_cat, nullptr, nullptr, lw2, lin2_b,
                h, HH, 0, jk_cat, 3 * HH, i * HH, jk_sum, scp,
                NN, 3 * HH, i > 0 ? 1 : 0);
  }

  // ---- layer aggregation + merge ----
  launch_gemm(EPI_LA, ASRC_PLAIN, G, stream,
              jk_cat, nullptr, nullptr, lawT, la_cat_b,
              merge, HH, 0, nullptr, 0, 0, jk_sum, la_w_d,
              NN, 3 * HH, 0);

  // ---- classifier ----
  launch_gemm(EPI_RELU, ASRC_PLAIN, G, stream,
              merge, nullptr, nullptr, c1wT, clf_b1,
              m1, HH, 0, nullptr, 0, 0, nullptr, nullptr,
              NN, HH, 0);
  clf2_kernel<<<(NN + 255) / 256, 256, 0, stream>>>(m1, clf_W2, clf_b2,
                                                    (float*)d_out, NN);
}
